// MultiViewGraphAttention_40785009443414
// MI455X (gfx1250) — compile-verified, matrix operand order checked by reading, not run on hardware
//
#include <hip/hip_runtime.h>
#include <hip/hip_bf16.h>

#define N_NODES 20000
#define N_EDGES 320000
#define E_TOT   (N_EDGES + N_NODES)   // edges + self loops
#define IN_DIM  256
#define HID     128
#define HEADS   4
#define HHID    (HEADS * HID)         // 512

typedef __attribute__((ext_vector_type(8)))  _Float16 v8h;
typedef __attribute__((ext_vector_type(16))) _Float16 v16h;
typedef __attribute__((ext_vector_type(8)))  float    v8f;

// ---------------------------------------------------------------------------
// monotone float<->uint mapping for atomic float max via atomicMax(u32)
__device__ __forceinline__ unsigned fmap(float f) {
    unsigned b = __float_as_uint(f);
    return (b & 0x80000000u) ? ~b : (b | 0x80000000u);
}
__device__ __forceinline__ float funmap(unsigned u) {
    unsigned b = (u & 0x80000000u) ? (u & 0x7FFFFFFFu) : ~u;
    return __uint_as_float(b);
}

// native GLOBAL_ATOMIC_ADD_F32 (avoid CAS-loop lowering)
__device__ __forceinline__ void atomAddF32(float* p, float v) {
    unsafeAtomicAdd(p, v);
}

// ---------------------------------------------------------------------------
__global__ void k_f32_to_f16(const float* __restrict__ in, _Float16* __restrict__ out, long n) {
    long t = (long)blockIdx.x * blockDim.x + threadIdx.x;
    if (t < n) out[t] = (_Float16)in[t];
}

// W[K,Nn] f32 -> Wt[Nn,K] f16 (transpose so B fragments are contiguous loads)
__global__ void k_transpose_f16(const float* __restrict__ W, _Float16* __restrict__ Wt,
                                int K, int Nn) {
    long t = (long)blockIdx.x * blockDim.x + threadIdx.x;
    if (t >= (long)K * Nn) return;
    int k = (int)(t / Nn), n = (int)(t % Nn);
    Wt[(long)n * K + k] = (_Float16)W[t];
}

__global__ void k_init_nh(unsigned* __restrict__ mmax, float* __restrict__ denom, int n) {
    int t = blockIdx.x * blockDim.x + threadIdx.x;
    if (t < n) { mmax[t] = 0u; denom[t] = 0.f; }   // 0 == mapped(-inf)
}

__global__ void k_zero_f32(float* __restrict__ p, long n) {
    long t = (long)blockIdx.x * blockDim.x + threadIdx.x;
    if (t < n) p[t] = 0.f;
}

// ---------------------------------------------------------------------------
// WMMA GEMM: C[M,Nn] f32 = A[M,K] f16 (row major) * Bt[Nn,K] f16 (B transposed)
// block = 256 threads = 8 waves; wave w owns NT n-tiles: n0 = (w + 8*i)*16.
// A fragment loaded once per K-step and reused for NT WMMAs (cuts A traffic NT-x).
template<int NT>
__global__ __launch_bounds__(256)
void k_gemm_wmma_f16(const _Float16* __restrict__ A, const _Float16* __restrict__ Bt,
                     float* __restrict__ C, int M, int K, int Nn) {
    const int lane = threadIdx.x & 31;
    const int wave = threadIdx.x >> 5;
    const int m0 = blockIdx.x * 16;
    const int r  = lane & 15;
    const int hf = lane >> 4;                  // 0 or 1 (lane half)
    const _Float16* arow = A + (long)(m0 + r) * K;

    v8f acc[NT];
#pragma unroll
    for (int i = 0; i < NT; ++i) acc[i] = (v8f){};

    for (int k0 = 0; k0 < K; k0 += 32) {
        v8h a0 = *(const v8h*)(arow + k0 + hf * 8);
        v8h a1 = *(const v8h*)(arow + k0 + 16 + hf * 8);
        v16h av = __builtin_shufflevector(a0, a1, 0,1,2,3,4,5,6,7,8,9,10,11,12,13,14,15);
#pragma unroll
        for (int i = 0; i < NT; ++i) {
            const int n0 = (wave + 8 * i) * 16;
            const _Float16* brow = Bt + (long)(n0 + r) * K;
            v8h b0 = *(const v8h*)(brow + k0 + hf * 8);
            v8h b1 = *(const v8h*)(brow + k0 + 16 + hf * 8);
            v16h bv = __builtin_shufflevector(b0, b1, 0,1,2,3,4,5,6,7,8,9,10,11,12,13,14,15);
            acc[i] = __builtin_amdgcn_wmma_f32_16x16x32_f16(false, av, false, bv,
                                                            (short)0, acc[i], false, false);
        }
    }
    // D layout: VGPR v -> row m0 + v + 8*hf, col n0 + r
#pragma unroll
    for (int i = 0; i < NT; ++i) {
        const int n0 = (wave + 8 * i) * 16;
        float* crow = C + (long)(m0 + hf * 8) * Nn + n0 + r;
#pragma unroll
        for (int v = 0; v < 8; ++v) crow[(long)v * Nn] = acc[i][v];
    }
}

// ---------------------------------------------------------------------------
// per-(node,head) attention logits: s_src/s_dst dot products over 128 channels
__global__ void k_node_scores(const float* __restrict__ hfeat,
                              const float* __restrict__ att_src,
                              const float* __restrict__ att_dst,
                              float* __restrict__ s_src, float* __restrict__ s_dst) {
    int t = blockIdx.x * blockDim.x + threadIdx.x;      // over N*HEADS
    if (t >= N_NODES * HEADS) return;
    int n = t >> 2, h = t & 3;
    const float* hp = hfeat + (long)n * HHID + h * HID;
    const float* as = att_src + h * HID;
    const float* ad = att_dst + h * HID;
    float ss = 0.f, sd = 0.f;
#pragma unroll 4
    for (int c = 0; c < HID; ++c) { float v = hp[c]; ss += v * as[c]; sd += v * ad[c]; }
    s_src[t] = ss; s_dst[t] = sd;
}

// phase A: edge logits (leaky relu) + segment max over dst (atomic u32 max)
__global__ void k_edge_max(const int* __restrict__ src, const int* __restrict__ dst,
                           const float* __restrict__ s_src, const float* __restrict__ s_dst,
                           float* __restrict__ escore, unsigned* __restrict__ mmax) {
    int e = blockIdx.x * blockDim.x + threadIdx.x;
    if (e >= E_TOT) return;
    int s = (e < N_EDGES) ? src[e] : (e - N_EDGES);
    int d = (e < N_EDGES) ? dst[e] : (e - N_EDGES);
#pragma unroll
    for (int h = 0; h < HEADS; ++h) {
        float v = s_src[s * HEADS + h] + s_dst[d * HEADS + h];
        v = (v > 0.f) ? v : 0.2f * v;                    // leaky_relu, slope 0.2
        escore[(long)e * HEADS + h] = v;
        atomicMax(&mmax[d * HEADS + h], fmap(v));
    }
}

// phase B: ex = exp(e - m[dst]); segment sum over dst; escore := ex
__global__ void k_edge_expsum(const int* __restrict__ dst,
                              const unsigned* __restrict__ mmax,
                              float* __restrict__ escore, float* __restrict__ denom) {
    int e = blockIdx.x * blockDim.x + threadIdx.x;
    if (e >= E_TOT) return;
    int d = (e < N_EDGES) ? dst[e] : (e - N_EDGES);
#pragma unroll
    for (int h = 0; h < HEADS; ++h) {
        float m  = funmap(mmax[d * HEADS + h]);
        float ex = __expf(escore[(long)e * HEADS + h] - m);
        escore[(long)e * HEADS + h] = ex;
        atomAddF32(&denom[d * HEADS + h], ex);
    }
}

// phase B2: normalize once: escore := ex / (denom[dst] + eps)  (kills per-channel divides)
__global__ void k_edge_alpha(const int* __restrict__ dst,
                             const float* __restrict__ denom, float* __restrict__ escore) {
    int t = blockIdx.x * blockDim.x + threadIdx.x;      // over E_TOT*HEADS
    if (t >= E_TOT * HEADS) return;
    int e = t >> 2, h = t & 3;
    int d = (e < N_EDGES) ? dst[e] : (e - N_EDGES);
    escore[t] = escore[t] / (denom[d * HEADS + h] + 1e-16f);
}

// phase C: out[dst] += alpha * h[src]; 128 lanes per edge (one per channel), 4 heads/lane
__global__ void k_edge_aggregate(const int* __restrict__ src, const int* __restrict__ dst,
                                 const float* __restrict__ alpha,
                                 const float* __restrict__ hfeat,
                                 float* __restrict__ outacc) {
    long t = (long)blockIdx.x * blockDim.x + threadIdx.x;
    long e = t >> 7;
    int  c = (int)(t & 127);
    if (e >= E_TOT) return;
    int s = (e < N_EDGES) ? src[e] : (int)(e - N_EDGES);
    int d = (e < N_EDGES) ? dst[e] : (int)(e - N_EDGES);
#pragma unroll
    for (int h = 0; h < HEADS; ++h) {
        atomAddF32(&outacc[(long)d * HHID + h * HID + c],
                   alpha[e * HEADS + h] * hfeat[(long)s * HHID + h * HID + c]);
    }
}

// mean over heads + bias + ELU; also emits f16 copy for the next WMMA GEMM
__global__ void k_mean_bias_elu(const float* __restrict__ outacc,
                                const float* __restrict__ bias,
                                float* __restrict__ hact, _Float16* __restrict__ hact_h) {
    int t = blockIdx.x * blockDim.x + threadIdx.x;      // over N*HID
    if (t >= N_NODES * HID) return;
    int n = t >> 7, c = t & 127;
    const float* o = outacc + (long)n * HHID;
    float a = 0.25f * (o[c] + o[HID + c] + o[2 * HID + c] + o[3 * HID + c]) + bias[c];
    a = (a > 0.f) ? a : expm1f(a);                      // ELU(alpha=1)
    hact[t] = a;
    hact_h[t] = (_Float16)a;
}

__global__ void k_add_bias_out(const float* __restrict__ Cin, const float* __restrict__ bias,
                               float* __restrict__ out) {
    int t = blockIdx.x * blockDim.x + threadIdx.x;      // over N*HID
    if (t >= N_NODES * HID) return;
    out[t] = Cin[t] + bias[t & 127];
}

// ---------------------------------------------------------------------------
extern "C" void kernel_launch(void* const* d_in, const int* in_sizes, int n_in,
                              void* d_out, int out_size, void* d_ws, size_t ws_size,
                              hipStream_t stream) {
    const float* x        = (const float*)d_in[0];
    const int*   src      = (const int*)d_in[1];            // edge_index[0]
    const int*   dst      = ((const int*)d_in[1]) + N_EDGES;// edge_index[1]
    const float* W1       = (const float*)d_in[2];
    const float* att_src1 = (const float*)d_in[3];
    const float* att_dst1 = (const float*)d_in[4];
    const float* b1       = (const float*)d_in[5];
    const float* W2       = (const float*)d_in[6];
    const float* att_src2 = (const float*)d_in[7];
    const float* att_dst2 = (const float*)d_in[8];
    const float* b2       = (const float*)d_in[9];
    const float* W_out    = (const float*)d_in[10];
    const float* b_out    = (const float*)d_in[11];
    float* out = (float*)d_out;

    // workspace carve-out (bytes)
    char* ws = (char*)d_ws;
    float*    hfeat  = (float*)   (ws);                 // N*512 f32      40,960,000
    float*    outacc = (float*)   (ws +  40960000);     // N*512 f32      40,960,000
    float*    ssrc   = (float*)   (ws +  81920000);     // N*4 f32           320,000
    float*    sdst   = (float*)   (ws +  82240000);     // N*4 f32           320,000
    unsigned* mmax   = (unsigned*)(ws +  82560000);     // N*4 u32           320,000
    float*    denom  = (float*)   (ws +  82880000);     // N*4 f32           320,000
    float*    esc    = (float*)   (ws +  83200000);     // Etot*4 f32      5,440,000
    float*    hact   = (float*)   (ws +  88640000);     // N*128 f32      10,240,000
    _Float16* xh     = (_Float16*)(ws +  98880000);     // N*256 f16      10,240,000
    _Float16* hacth  = (_Float16*)(ws + 109120000);     // N*128 f16       5,120,000
    _Float16* wt1    = (_Float16*)(ws + 114240000);     // 512*256 f16       262,144
    _Float16* wt2    = (_Float16*)(ws + 114502144);     // 512*128 f16       131,072
    _Float16* wto    = (_Float16*)(ws + 114633216);     // 128*128 f16        32,768

    const int B = 256;
    auto blk = [](long n, int b) { return (unsigned)((n + b - 1) / b); };

    // operand conversion / weight transposes (B pre-transposed for WMMA fragments)
    k_f32_to_f16   <<<blk((long)N_NODES * IN_DIM, B), B, 0, stream>>>(x, xh, (long)N_NODES * IN_DIM);
    k_transpose_f16<<<blk((long)IN_DIM * HHID, B), B, 0, stream>>>(W1, wt1, IN_DIM, HHID);
    k_transpose_f16<<<blk((long)HID * HHID, B), B, 0, stream>>>(W2, wt2, HID, HHID);
    k_transpose_f16<<<blk((long)HID * HID, B), B, 0, stream>>>(W_out, wto, HID, HID);

    // ---------------- layer 1 ----------------
    k_init_nh <<<blk(N_NODES * HEADS, B), B, 0, stream>>>(mmax, denom, N_NODES * HEADS);
    k_zero_f32<<<blk((long)N_NODES * HHID, B), B, 0, stream>>>(outacc, (long)N_NODES * HHID);

    // h = x @ W1 : M=20000, K=256, Nn=512 ; 8 waves x 4 n-tiles = 512 cols per block
    k_gemm_wmma_f16<4><<<dim3(N_NODES / 16, 1), B, 0, stream>>>(xh, wt1, hfeat,
                                                                N_NODES, IN_DIM, HHID);
    k_node_scores<<<blk(N_NODES * HEADS, B), B, 0, stream>>>(hfeat, att_src1, att_dst1, ssrc, sdst);
    k_edge_max      <<<blk(E_TOT, B), B, 0, stream>>>(src, dst, ssrc, sdst, esc, mmax);
    k_edge_expsum   <<<blk(E_TOT, B), B, 0, stream>>>(dst, mmax, esc, denom);
    k_edge_alpha    <<<blk((long)E_TOT * HEADS, B), B, 0, stream>>>(dst, denom, esc);
    k_edge_aggregate<<<blk((long)E_TOT * HID, B), B, 0, stream>>>(src, dst, esc, hfeat, outacc);
    k_mean_bias_elu <<<blk((long)N_NODES * HID, B), B, 0, stream>>>(outacc, b1, hact, hacth);

    // ---------------- layer 2 ----------------
    k_init_nh <<<blk(N_NODES * HEADS, B), B, 0, stream>>>(mmax, denom, N_NODES * HEADS);
    k_zero_f32<<<blk((long)N_NODES * HHID, B), B, 0, stream>>>(outacc, (long)N_NODES * HHID);

    // h = hact @ W2 : M=20000, K=128, Nn=512
    k_gemm_wmma_f16<4><<<dim3(N_NODES / 16, 1), B, 0, stream>>>(hacth, wt2, hfeat,
                                                                N_NODES, HID, HHID);
    k_node_scores<<<blk(N_NODES * HEADS, B), B, 0, stream>>>(hfeat, att_src2, att_dst2, ssrc, sdst);
    k_edge_max      <<<blk(E_TOT, B), B, 0, stream>>>(src, dst, ssrc, sdst, esc, mmax);
    k_edge_expsum   <<<blk(E_TOT, B), B, 0, stream>>>(dst, mmax, esc, denom);
    k_edge_alpha    <<<blk((long)E_TOT * HEADS, B), B, 0, stream>>>(dst, denom, esc);
    k_edge_aggregate<<<blk((long)E_TOT * HID, B), B, 0, stream>>>(src, dst, esc, hfeat, outacc);
    k_mean_bias_elu <<<blk((long)N_NODES * HID, B), B, 0, stream>>>(outacc, b2, hact, hacth);

    // ---------------- output projection ----------------
    // hfeat (as [N,128], row stride 128) = hact @ W_out : M=20000, K=128, Nn=128 ; 8 waves x 1 tile = 128 cols
    k_gemm_wmma_f16<1><<<dim3(N_NODES / 16, 1), B, 0, stream>>>(hacth, wto, hfeat,
                                                                N_NODES, HID, HID);
    k_add_bias_out<<<blk((long)N_NODES * HID, B), B, 0, stream>>>(hfeat, b_out, out);
}
